// GNNReg_67336497266975
// MI455X (gfx1250) — matrix operand order repaired and checked by reading, not compiled, not run on hardware
//
#include <hip/hip_runtime.h>
#include <hip/hip_bf16.h>

// ---------------------------------------------------------------------------
// GNN mixture-property pipeline for gfx1250 (MI455X).
// Memory-bound (~1.3GB HBM traffic vs ~80 GFLOP) -> fuse everything, run all
// big GEMMs on v_wmma_f32_16x16x32_f16 (f32 accumulate), stage the GINE hidden
// layer in LDS, use relu-gated f32 global atomics for scatter/pool.
// ---------------------------------------------------------------------------

#define Nn   250000   // atoms per molecule batch  (250000 = 15625 * 16)
#define Ee   500000   // directed bonds            (500000 = 31250 * 16)
#define Bm   8192     // mixtures
#define DIM  128
#define NF   64
#define NEF  8

typedef __attribute__((ext_vector_type(16))) _Float16 v16h;
typedef __attribute__((ext_vector_type(8)))  float    v8f;

__device__ __forceinline__ float frelu(float x) { return x > 0.f ? x : 0.f; }

__device__ __forceinline__ v8f wmma_f16(v16h a, v16h b, v8f c) {
    // (neg_a, A, neg_b, B, c_mod, C, reuse_a, reuse_b)
    return __builtin_amdgcn_wmma_f32_16x16x32_f16(false, a, false, b,
                                                  (short)0, c, false, false);
}

// ---- A fragment (16x32 f16, MxK) from a row-major f32 matrix --------------
// lanes 0-15: row=lane,    K slots {kOff+0..7,  kOff+16..23}
// lanes 16-31: row=lane-16, K slots {kOff+8..15, kOff+24..31}
__device__ __forceinline__ v16h load_a_f32(const float* __restrict__ src, int ld,
                                           int row0, int lane, int kOff) {
    int r   = row0 + (lane & 15);
    int hlf = lane >> 4;
    const float* p = src + (size_t)r * ld + kOff + hlf * 8;
    v16h a;
#pragma unroll
    for (int i = 0; i < 8; ++i) {
        a[i]     = (_Float16)p[i];
        a[8 + i] = (_Float16)p[16 + i];
    }
    return a;
}

// same, but element-wise sum of two f32 matrices (h + aggr)
__device__ __forceinline__ v16h load_a_sum_f32(const float* __restrict__ s0,
                                               const float* __restrict__ s1, int ld,
                                               int row0, int lane, int kOff) {
    int r   = row0 + (lane & 15);
    int hlf = lane >> 4;
    size_t o = (size_t)r * ld + kOff + hlf * 8;
    const float* p = s0 + o;
    const float* q = s1 + o;
    v16h a;
#pragma unroll
    for (int i = 0; i < 8; ++i) {
        a[i]     = (_Float16)(p[i] + q[i]);
        a[8 + i] = (_Float16)(p[16 + i] + q[16 + i]);
    }
    return a;
}

// A fragment from LDS row-major _Float16 [16][256]
__device__ __forceinline__ v16h load_a_lds(const _Float16* __restrict__ lds,
                                           int lane, int kt) {
    int r   = lane & 15;
    int hlf = lane >> 4;
    const _Float16* p = lds + r * 256 + kt * 32 + hlf * 8;
    v16h a;
#pragma unroll
    for (int i = 0; i < 8; ++i) {
        a[i]     = p[i];
        a[8 + i] = p[16 + i];
    }
    return a;
}

// ---- B fragment: pre-packed so each lane reads 16 contiguous halves -------
// pack[((ct*ktiles+kt)*32 + lane)*16 + i] = W[kt*32 + (lane/16)*8 + (i%8) + (i/8)*16][ct*16 + lane%16]
__device__ __forceinline__ v16h load_b_pack(const _Float16* __restrict__ pack,
                                            int ktiles, int kt, int ct, int lane) {
    const _Float16* p = pack + ((size_t)((ct * ktiles + kt) * 32 + lane) << 4);
    return *(const v16h*)p;   // 32B aligned, fully coalesced
}

// ---------------------------------------------------------------------------
// K0: convert f32 weight [K, Ncols] -> f16 B-fragment-packed layout
// ---------------------------------------------------------------------------
__global__ void pack_b_kernel(const float* __restrict__ w, _Float16* __restrict__ out,
                              int K, int Ncols) {
    int idx   = blockIdx.x * 256 + threadIdx.x;
    int total = K * Ncols;
    if (idx >= total) return;
    int ktiles = K >> 5;
    int i  = idx & 15;
    int ln = (idx >> 4) & 31;
    int kt = (idx >> 9) % ktiles;
    int ct = idx / (512 * ktiles);
    int k   = kt * 32 + (ln >> 4) * 8 + (i & 7) + (i >> 3) * 16;
    int col = ct * 16 + (ln & 15);
    out[idx] = (_Float16)w[k * Ncols + col];
}

// ---------------------------------------------------------------------------
// K1: h = relu(x @ W0 + b0)   x:[N,64]  W0:[64,128]  -> h:[N,128]
// one wave per 16-row tile; 8 col-tiles x 2 k-steps of WMMA
// ---------------------------------------------------------------------------
__global__ void __launch_bounds__(128)
lin0_kernel(const float* __restrict__ x, const _Float16* __restrict__ w0p,
            const float* __restrict__ bias, float* __restrict__ h, int nTiles) {
    int wave = threadIdx.x >> 5;
    int lane = threadIdx.x & 31;
    int tile = blockIdx.x * 4 + wave;
    if (tile >= nTiles) return;            // wave-uniform: EXEC all-ones below
    int row0 = tile * 16;

    for (int ct = 0; ct < 8; ++ct) {
        v8f acc = {};
#pragma unroll
        for (int kt = 0; kt < 2; ++kt) {
            v16h a = load_a_f32(x, NF, row0, lane, kt * 32);
            v16h b = load_b_pack(w0p, 2, kt, ct, lane);
            acc = wmma_f16(a, b, acc);
        }
        int col   = ct * 16 + (lane & 15);
        float bv  = bias[col];
        int mbase = (lane >> 4) * 8;
#pragma unroll
        for (int r = 0; r < 8; ++r) {
            int m = row0 + mbase + r;
            h[(size_t)m * DIM + col] = frelu(acc[r] + bv);
        }
    }
}

// ---------------------------------------------------------------------------
// K2: fused edge transform + message + scatter:
//     aggr[dst] += relu(h[src] + ea @ Wt + bt)
// one thread per (edge, channel); relu-gated f32 global atomics
// ---------------------------------------------------------------------------
__global__ void __launch_bounds__(256)
edge_kernel(const float* __restrict__ h, const int* __restrict__ ei,
            const float* __restrict__ ea, const float* __restrict__ tw,
            const float* __restrict__ tb, float* __restrict__ aggr) {
    long long idx = (long long)blockIdx.x * 256 + threadIdx.x;
    long long total = (long long)Ee * DIM;
    if (idx >= total) return;
    int e = (int)(idx >> 7);
    int c = (int)(idx & (DIM - 1));
    int src = ei[e];        // edge_index[0][e]
    int dst = ei[Ee + e];   // edge_index[1][e]
    float eat = tb[c];
#pragma unroll
    for (int k = 0; k < NEF; ++k)
        eat += ea[e * NEF + k] * tw[k * DIM + c];
    float msg = h[(size_t)src * DIM + c] + eat;
    if (msg > 0.f)
        atomicAdd(&aggr[(size_t)dst * DIM + c], msg);
}

// ---------------------------------------------------------------------------
// K3: fused GINE MLP + pooling
//     out = relu( relu((h+aggr)@G1 + b1) @ G2 + b2 );  fp[batch[i]] += out[i]
// one wave per 16-row tile; hidden [16,256] f16 staged in per-wave LDS
// (no __syncthreads: same-wave DS ops are in-order, arrays are wave-private)
// ---------------------------------------------------------------------------
__global__ void __launch_bounds__(128)
gine_kernel(const float* __restrict__ h, const float* __restrict__ aggr,
            const _Float16* __restrict__ g1p, const float* __restrict__ gb1,
            const _Float16* __restrict__ g2p, const float* __restrict__ gb2,
            const int* __restrict__ batch, float* __restrict__ fp, int nTiles) {
    __shared__ _Float16 smid[4][16 * 256];   // 32 KB / block
    int wave = threadIdx.x >> 5;
    int lane = threadIdx.x & 31;
    int tile = blockIdx.x * 4 + wave;
    if (tile >= nTiles) return;
    int row0 = tile * 16;
    _Float16* mid = smid[wave];

    // stage 1: mid = relu((h+aggr) @ G1 + b1)   [16,256]
    for (int ct = 0; ct < 16; ++ct) {
        v8f acc = {};
#pragma unroll
        for (int kt = 0; kt < 4; ++kt) {
            v16h a = load_a_sum_f32(h, aggr, DIM, row0, lane, kt * 32);
            v16h b = load_b_pack(g1p, 4, kt, ct, lane);
            acc = wmma_f16(a, b, acc);
        }
        int col   = ct * 16 + (lane & 15);
        float bv  = gb1[col];
        int mbase = (lane >> 4) * 8;
#pragma unroll
        for (int r = 0; r < 8; ++r)
            mid[(mbase + r) * 256 + col] = (_Float16)frelu(acc[r] + bv);
    }

    // stage 2: out = relu(mid @ G2 + b2); pool into fp by batch id
    for (int ct = 0; ct < 8; ++ct) {
        v8f acc = {};
#pragma unroll
        for (int kt = 0; kt < 8; ++kt) {
            v16h a = load_a_lds(mid, lane, kt);
            v16h b = load_b_pack(g2p, 8, kt, ct, lane);
            acc = wmma_f16(a, b, acc);
        }
        int col   = ct * 16 + (lane & 15);
        float bv  = gb2[col];
        int mbase = (lane >> 4) * 8;
#pragma unroll
        for (int r = 0; r < 8; ++r) {
            int node = row0 + mbase + r;
            float v  = frelu(acc[r] + bv);
            int bb   = batch[node];
            atomicAdd(&fp[(size_t)bb * DIM + col], v);
        }
    }
}

// ---------------------------------------------------------------------------
// K4: mixture 2-node GINE + MLP head, one block (128 threads) per mixture
// ---------------------------------------------------------------------------
__global__ void __launch_bounds__(128)
mixture_kernel(const float* __restrict__ fp1, const float* __restrict__ fp2,
               const float* __restrict__ ratio1, const float* __restrict__ ratio2,
               const float* __restrict__ temp, const float* __restrict__ h_inter,
               const float* __restrict__ h_intra1,
               const float* __restrict__ mt_w, const float* __restrict__ mt_b,
               const float* __restrict__ m_w1, const float* __restrict__ m_b1,
               const float* __restrict__ m_w2, const float* __restrict__ m_b2,
               const float* __restrict__ fc1_w, const float* __restrict__ fc1_b,
               const float* __restrict__ fc2_w, const float* __restrict__ fc2_b,
               const float* __restrict__ fc3_w, const float* __restrict__ fc3_b,
               float* __restrict__ out) {
    __shared__ float t0[DIM], t1[DIM], mid0[2 * DIM], mid1[2 * DIM];
    __shared__ float g[DIM], hh[DIM + 1], h2s[DIM], red[DIM];
    int b = blockIdx.x;
    int c = threadIdx.x;

    float r1 = ratio1[b], r2 = ratio2[b];
    bool pure = (r2 == 0.0f);
    float n0 = fp1[(size_t)b * DIM + c] * r1;
    float n1 = pure ? n0 : fp2[(size_t)b * DIM + c] * r2;
    float er = pure ? h_intra1[b] : h_inter[b];
    float ef = er * mt_w[c] + mt_b[c];
    t0[c] = n0 + frelu(n1 + ef);
    t1[c] = n1 + frelu(n0 + ef);
    __syncthreads();

    // mnn hidden: 256 outputs, 2 per thread, both graph nodes together
#pragma unroll
    for (int jj = 0; jj < 2; ++jj) {
        int j = c + jj * DIM;
        float s0 = m_b1[j], s1 = s0;
        for (int k = 0; k < DIM; ++k) {
            float w = m_w1[k * 2 * DIM + j];
            s0 += t0[k] * w;
            s1 += t1[k] * w;
        }
        mid0[j] = frelu(s0);
        mid1[j] = frelu(s1);
    }
    __syncthreads();

    // fp = relu(mnn(t0)) + relu(mnn(t1))
    {
        float s0 = m_b2[c], s1 = s0;
        for (int j = 0; j < 2 * DIM; ++j) {
            float w = m_w2[j * DIM + c];
            s0 += mid0[j] * w;
            s1 += mid1[j] * w;
        }
        g[c] = frelu(s0) + frelu(s1);
    }
    __syncthreads();

    // fc1 + concat(10*temp)
    {
        float s = fc1_b[c];
        for (int k = 0; k < DIM; ++k) s += g[k] * fc1_w[k * DIM + c];
        hh[c] = frelu(s);
        if (c == 0) hh[DIM] = 10.0f * temp[b];
    }
    __syncthreads();

    // fc2 over 129 inputs
    {
        float s = fc2_b[c];
        for (int k = 0; k < DIM + 1; ++k) s += hh[k] * fc2_w[k * DIM + c];
        h2s[c] = frelu(s);
    }
    __syncthreads();

    // fc3 reduction -> scalar
    red[c] = h2s[c] * fc3_w[c];
    __syncthreads();
    for (int s = 64; s > 0; s >>= 1) {
        if (c < s) red[c] += red[c + s];
        __syncthreads();
    }
    if (c == 0) out[b] = red[0] + fc3_b[0];
}

// ---------------------------------------------------------------------------
extern "C" void kernel_launch(void* const* d_in, const int* in_sizes, int n_in,
                              void* d_out, int out_size, void* d_ws, size_t ws_size,
                              hipStream_t stream) {
    // inputs in setup_inputs() dict order; params flattened in insertion order
    const float* x1       = (const float*)d_in[0];
    const int*   ei1      = (const int*)  d_in[1];
    const float* ea1      = (const float*)d_in[2];
    const int*   batch1   = (const int*)  d_in[3];
    const float* ratio1   = (const float*)d_in[4];
    const float* x2       = (const float*)d_in[5];
    const int*   ei2      = (const int*)  d_in[6];
    const float* ea2      = (const float*)d_in[7];
    const int*   batch2   = (const int*)  d_in[8];
    const float* ratio2   = (const float*)d_in[9];
    const float* temp     = (const float*)d_in[10];
    const float* h_inter  = (const float*)d_in[11];
    const float* h_intra1 = (const float*)d_in[12];
    const float* lin0_w   = (const float*)d_in[13];
    const float* lin0_b   = (const float*)d_in[14];
    const float* trans_w  = (const float*)d_in[15];
    const float* trans_b  = (const float*)d_in[16];
    const float* g_w1     = (const float*)d_in[17];
    const float* g_b1     = (const float*)d_in[18];
    const float* g_w2     = (const float*)d_in[19];
    const float* g_b2     = (const float*)d_in[20];
    const float* mt_w     = (const float*)d_in[21];
    const float* mt_b     = (const float*)d_in[22];
    const float* m_w1     = (const float*)d_in[23];
    const float* m_b1     = (const float*)d_in[24];
    const float* m_w2     = (const float*)d_in[25];
    const float* m_b2     = (const float*)d_in[26];
    const float* fc1_w    = (const float*)d_in[27];
    const float* fc1_b    = (const float*)d_in[28];
    const float* fc2_w    = (const float*)d_in[29];
    const float* fc2_b    = (const float*)d_in[30];
    const float* fc3_w    = (const float*)d_in[31];
    const float* fc3_b    = (const float*)d_in[32];

    // workspace carve-up (all 256B aligned)
    char* ws = (char*)d_ws;
    float* h    = (float*)ws;                 ws += (size_t)Nn * DIM * 4;   // 128 MB
    float* aggr = (float*)ws;                 ws += (size_t)Nn * DIM * 4;   // 128 MB
    float* fp1  = (float*)ws;                 ws += (size_t)Bm * DIM * 4;   // 4 MB
    float* fp2  = (float*)ws;                 ws += (size_t)Bm * DIM * 4;   // 4 MB
    _Float16* w0p = (_Float16*)ws;            ws += (size_t)NF * DIM * 2;
    _Float16* g1p = (_Float16*)ws;            ws += (size_t)DIM * 2 * DIM * 2;
    _Float16* g2p = (_Float16*)ws;            ws += (size_t)2 * DIM * DIM * 2;

    // K0: pack weights into f16 B-fragment layout
    pack_b_kernel<<<(NF * DIM + 255) / 256, 256, 0, stream>>>(lin0_w, w0p, NF, DIM);
    pack_b_kernel<<<(DIM * 2 * DIM + 255) / 256, 256, 0, stream>>>(g_w1, g1p, DIM, 2 * DIM);
    pack_b_kernel<<<(2 * DIM * DIM + 255) / 256, 256, 0, stream>>>(g_w2, g2p, 2 * DIM, DIM);

    hipMemsetAsync(fp1, 0, (size_t)Bm * DIM * 4, stream);
    hipMemsetAsync(fp2, 0, (size_t)Bm * DIM * 4, stream);

    const int nTiles   = Nn / 16;                 // 15625
    const int gemmGrid = (nTiles + 3) / 4;
    const long long edgeThreads = (long long)Ee * DIM;
    const int edgeGrid = (int)((edgeThreads + 255) / 256);

    // ---- molecule 1 ----
    hipMemsetAsync(aggr, 0, (size_t)Nn * DIM * 4, stream);
    lin0_kernel<<<gemmGrid, 128, 0, stream>>>(x1, w0p, lin0_b, h, nTiles);
    edge_kernel<<<edgeGrid, 256, 0, stream>>>(h, ei1, ea1, trans_w, trans_b, aggr);
    gine_kernel<<<gemmGrid, 128, 0, stream>>>(h, aggr, g1p, g_b1, g2p, g_b2,
                                              batch1, fp1, nTiles);

    // ---- molecule 2 ----
    hipMemsetAsync(aggr, 0, (size_t)Nn * DIM * 4, stream);
    lin0_kernel<<<gemmGrid, 128, 0, stream>>>(x2, w0p, lin0_b, h, nTiles);
    edge_kernel<<<edgeGrid, 256, 0, stream>>>(h, ei2, ea2, trans_w, trans_b, aggr);
    gine_kernel<<<gemmGrid, 128, 0, stream>>>(h, aggr, g1p, g_b1, g2p, g_b2,
                                              batch2, fp2, nTiles);

    // ---- mixture graph + head ----
    mixture_kernel<<<Bm, 128, 0, stream>>>(fp1, fp2, ratio1, ratio2, temp,
                                           h_inter, h_intra1,
                                           mt_w, mt_b, m_w1, m_b1, m_w2, m_b2,
                                           fc1_w, fc1_b, fc2_w, fc2_b,
                                           fc3_w, fc3_b, (float*)d_out);
}
